// GNN_LSTM_13039520710885
// MI455X (gfx1250) — matrix operand order repaired and checked by reading, not compiled, not run on hardware
//
#include <hip/hip_runtime.h>
#include <hip/hip_bf16.h>

typedef _Float16 v8h  __attribute__((ext_vector_type(8)));
typedef _Float16 v16h __attribute__((ext_vector_type(16)));
typedef float    v8f  __attribute__((ext_vector_type(8)));

#define N_NODES 512
#define F_IN_   512
#define HID     64
#define TSEQ    16
#define TTS     200
#define KSEL    76
#define GDIM    256      // 4*HID packed gate width
#define EPSF    1e-8f

__device__ __forceinline__ float sigf_(float x) { return 1.f / (1.f + expf(-x)); }

// ===========================================================================
// Fragment-packed layouts (wave32; this file's contiguous k order, same for A and B):
//  A-pack: element ((mt*KC + kc)*512 + lane*16 + j)
//     lane = r + 16*hi (r = row in tile), K = kc*32 + hi*16 + j
//  B-pack: element ((nt*KC + kc)*512 + lane*16 + j)
//     lane = n + 16*hi (n = col in tile), K = kc*32 + hi*16 + j
// Each lane's fragment = 16 contiguous halves = 32B -> b128-pair loads.
// ===========================================================================

// ---------------------------------------------------------------------------
// WMMA GEMM on packed operands. 32x32 C tile per wave (4 WMMAs / k-step).
// C written row-major f32 (Cf) and/or B-pack f16 (Chp) for GEMM chaining.
// blockIdx.z = batch with element strides on A, B, C.
// ---------------------------------------------------------------------------
__global__ __launch_bounds__(32)
void wmma_gemm_pk(const _Float16* __restrict__ Apk, const _Float16* __restrict__ Bpk,
                  float* __restrict__ Cf, _Float16* __restrict__ Chp,
                  int K, int N, int MC,
                  long long strideA, long long strideB, long long strideC)
{
    const int lane = threadIdx.x;
    const int KC   = K >> 5;
    const int mt0  = blockIdx.y << 1;   // two 16-row tiles
    const int nt0  = blockIdx.x << 1;   // two 16-col tiles

    const v16h* Af0 = (const v16h*)(Apk + (size_t)strideA * blockIdx.z) + (size_t)mt0 * KC * 32 + lane;
    const v16h* Af1 = Af0 + (size_t)KC * 32;
    const v16h* Bf0 = (const v16h*)(Bpk + (size_t)strideB * blockIdx.z) + (size_t)nt0 * KC * 32 + lane;
    const v16h* Bf1 = Bf0 + (size_t)KC * 32;

    v8f acc00 = {}, acc01 = {}, acc10 = {}, acc11 = {};
    for (int kc = 0; kc < KC; ++kc) {
        const size_t o = (size_t)kc << 5;      // kc*32 v16h units
        v16h a0 = Af0[o];
        v16h a1 = Af1[o];
        v16h b0 = Bf0[o];
        v16h b1 = Bf1[o];
        __builtin_prefetch((const void*)(Af0 + o + 32), 0, 1);
        __builtin_prefetch((const void*)(Bf0 + o + 32), 0, 1);
        acc00 = __builtin_amdgcn_wmma_f32_16x16x32_f16(false, a0, false, b0, (short)0, acc00, false, false);
        acc01 = __builtin_amdgcn_wmma_f32_16x16x32_f16(false, a0, false, b1, (short)0, acc01, false, false);
        acc10 = __builtin_amdgcn_wmma_f32_16x16x32_f16(false, a1, false, b0, (short)0, acc10, false, false);
        acc11 = __builtin_amdgcn_wmma_f32_16x16x32_f16(false, a1, false, b1, (short)0, acc11, false, false);
    }

    const int r  = lane & 15;
    const int hi = lane >> 4;

    if (Cf) {
        float* Cz = Cf + (size_t)strideC * blockIdx.z;
        #pragma unroll
        for (int ti = 0; ti < 2; ++ti) {
            const int m0 = (mt0 + ti) << 4;
            #pragma unroll
            for (int tj = 0; tj < 2; ++tj) {
                const int n0 = (nt0 + tj) << 4;
                const v8f acc = (ti == 0) ? (tj == 0 ? acc00 : acc01)
                                          : (tj == 0 ? acc10 : acc11);
                #pragma unroll
                for (int i = 0; i < 8; ++i)
                    Cz[(size_t)(m0 + (hi << 3) + i) * N + n0 + r] = acc[i];
            }
        }
    }
    if (Chp) {
        // write directly in B-pack layout of the consumer GEMM (K rows = M here)
        _Float16* Cz = Chp + (size_t)strideC * blockIdx.z;
        #pragma unroll
        for (int ti = 0; ti < 2; ++ti) {
            const int m0      = (mt0 + ti) << 4;
            const int kc2     = m0 >> 5;
            const int halfbit = (m0 >> 4) & 1;
            const int lane2   = r + (halfbit << 4);
            #pragma unroll
            for (int tj = 0; tj < 2; ++tj) {
                const int n0 = (nt0 + tj) << 4;
                const int nt = n0 >> 4;
                const v8f acc = (ti == 0) ? (tj == 0 ? acc00 : acc01)
                                          : (tj == 0 ? acc10 : acc11);
                v8h h8;
                #pragma unroll
                for (int i = 0; i < 8; ++i) h8[i] = (_Float16)acc[i];
                *(v8h*)(Cz + ((size_t)(nt * MC + kc2) << 9) + lane2 * 16 + (hi << 3)) = h8;
            }
        }
    }
}

// ---------------------------------------------------------------------------
// Dense normalized adjacency construction (E = N^2 random edges + self loops)
// ---------------------------------------------------------------------------
__global__ void k_deg(const int* __restrict__ ei, int E, float* __restrict__ deg) {
    int e = blockIdx.x * blockDim.x + threadIdx.x;
    if (e < E) atomicAdd(&deg[ei[E + e]], 1.f);
}
__global__ void k_dinv(const float* __restrict__ deg, float* __restrict__ dinv) {
    int i = blockIdx.x * blockDim.x + threadIdx.x;
    if (i < N_NODES) dinv[i] = rsqrtf(fmaxf(deg[i] + 1.f, 1.f)); // +1 self loop
}
__global__ void k_adj(const int* __restrict__ ei, int E,
                      const float* __restrict__ dinv, float* __restrict__ A) {
    int e = blockIdx.x * blockDim.x + threadIdx.x;
    if (e < E) {
        int s = ei[e], d = ei[E + e];
        atomicAdd(&A[(size_t)d * N_NODES + s], dinv[s] * dinv[d]);
    }
}
__global__ void k_selfloop(const float* __restrict__ dinv, float* __restrict__ A) {
    int i = blockIdx.x * blockDim.x + threadIdx.x;
    if (i < N_NODES) A[(size_t)i * N_NODES + i] += dinv[i] * dinv[i];
}

// ---------------------------------------------------------------------------
// Packing kernels (f32 row-major -> f16 fragment-packed)
// ---------------------------------------------------------------------------
// A-pack for fixed M=512, K=512 matrices; nbatch batches of 512*512
__global__ void k_packA(const float* __restrict__ src, _Float16* __restrict__ dst, int nbatch) {
    const int idx = blockIdx.x * blockDim.x + threadIdx.x;
    const int per = N_NODES * F_IN_;
    if (idx >= nbatch * per) return;
    const int b = idx / per, rem = idx - b * per;
    const int tile = rem >> 9, e = rem & 511, lane = e >> 4, j = e & 15;
    const int KC = F_IN_ >> 5;
    const int mt = tile / KC, kc = tile - mt * KC;
    const int r = lane & 15, hi = lane >> 4;
    const int k = kc * 32 + hi * 16 + j;
    const int m = (mt << 4) + r;
    dst[idx] = (_Float16)src[(size_t)b * per + (size_t)m * F_IN_ + k];
}
// B-pack of 4 gate weight matrices (rows K x 64 each) into packed (K x 256)
__global__ void k_pack4B(const float* __restrict__ W0, const float* __restrict__ W1,
                         const float* __restrict__ W2, const float* __restrict__ W3,
                         _Float16* __restrict__ out, int K) {
    const int idx = blockIdx.x * blockDim.x + threadIdx.x;
    if (idx >= K * GDIM) return;
    const int KC = K >> 5;
    const int tile = idx >> 9, e = idx & 511, lane = e >> 4, j = e & 15;
    const int nt = tile / KC, kc = tile - nt * KC;
    const int k   = kc * 32 + (lane >> 4) * 16 + j;
    const int col = (nt << 4) + (lane & 15);
    const int g = col >> 6, jj = col & 63;
    const float* W = (g == 0) ? W0 : (g == 1) ? W1 : (g == 2) ? W2 : W3;
    out[idx] = (_Float16)W[(size_t)k * HID + jj];
}

// A-pack address for the h matrix (M=512, K=64): row m, col k -> packed index
__device__ __forceinline__ size_t hpack_idx(int m, int k) {
    const int mt = m >> 4, r = m & 15;
    const int kc = k >> 5, kk = k & 31;
    int hi, j;
    hi = kk >> 4; j = kk & 15;
    return ((size_t)(mt * 2 + kc) << 9) + (size_t)(r + (hi << 4)) * 16 + j;
}

__global__ void k_inith(const float* __restrict__ h0, const float* __restrict__ c0,
                        float* __restrict__ hf, _Float16* __restrict__ hhpk,
                        float* __restrict__ cf) {
    int i = blockIdx.x * blockDim.x + threadIdx.x;
    if (i < N_NODES * HID) {
        float h = h0[i];
        hf[i] = h; cf[i] = c0[i];
        hhpk[hpack_idx(i >> 6, i & 63)] = (_Float16)h;
    }
}

// ---------------------------------------------------------------------------
// Gate nonlinearity + (h,c) update; writes h both f32 and A-packed f16
// ---------------------------------------------------------------------------
__global__ __launch_bounds__(256)
void k_gate(const float* __restrict__ AXWt, const float* __restrict__ AHW,
            const float* __restrict__ bxi, const float* __restrict__ bxf,
            const float* __restrict__ bxo, const float* __restrict__ bxm,
            const float* __restrict__ bhi, const float* __restrict__ bhf,
            const float* __restrict__ bho, const float* __restrict__ bhm,
            float* __restrict__ c, float* __restrict__ h, _Float16* __restrict__ hhpk)
{
    int idx = blockIdx.x * blockDim.x + threadIdx.x;   // N*H threads
    int nrow = idx >> 6, k = idx & 63;
    const float* ax = AXWt + (size_t)nrow * GDIM;
    const float* ah = AHW  + (size_t)nrow * GDIM;

    float gi = fmaxf(ax[0 * HID + k] + bxi[k], 0.f) + fmaxf(ah[0 * HID + k] + bhi[k], 0.f);
    float gf = fmaxf(ax[1 * HID + k] + bxf[k], 0.f) + fmaxf(ah[1 * HID + k] + bhf[k], 0.f);
    float go = fmaxf(ax[2 * HID + k] + bxo[k], 0.f) + fmaxf(ah[2 * HID + k] + bho[k], 0.f);
    float gm = fmaxf(ax[3 * HID + k] + bxm[k], 0.f) + fmaxf(ah[3 * HID + k] + bhm[k], 0.f);

    float ig = sigf_(gi), fg = sigf_(gf), og = sigf_(go), m = fmaxf(gm, 0.f);
    float cn = tanhf(ig * m + fg * c[idx]);
    float hn = og * tanhf(cn);
    c[idx] = cn; h[idx] = hn;
    hhpk[hpack_idx(nrow, k)] = (_Float16)hn;
}

// ---------------------------------------------------------------------------
// Scoring + standardize + sigmoid scale + bitonic top-K + pool loss
// ---------------------------------------------------------------------------
__global__ __launch_bounds__(512)
void k_pool(const float* __restrict__ h, const float* __restrict__ v,
            float* __restrict__ xscaled, float* __restrict__ newx,
            float* __restrict__ scalars)
{
    __shared__ float sv[512]; __shared__ int si[512];
    __shared__ float red[512]; __shared__ float vsh[64];
    __shared__ float sA, sB;
    int tid = threadIdx.x;

    if (tid < 64) vsh[tid] = v[tid];
    red[tid] = (tid < 64) ? v[tid] * v[tid] : 0.f;
    __syncthreads();
    for (int s = 256; s > 0; s >>= 1) { if (tid < s) red[tid] += red[tid + s]; __syncthreads(); }
    if (tid == 0) sA = sqrtf(red[0]) + EPSF;       // ||v|| + eps
    __syncthreads();

    float dot = 0.f;
    for (int k = 0; k < HID; ++k) dot += h[tid * HID + k] * vsh[k];
    float score = dot / sA;

    red[tid] = score; __syncthreads();
    for (int s = 256; s > 0; s >>= 1) { if (tid < s) red[tid] += red[tid + s]; __syncthreads(); }
    if (tid == 0) sB = red[0] / 512.f;             // mean
    __syncthreads();
    float mu = sB, d0 = score - mu;
    red[tid] = d0 * d0; __syncthreads();
    for (int s = 256; s > 0; s >>= 1) { if (tid < s) red[tid] += red[tid + s]; __syncthreads(); }
    if (tid == 0) sA = sqrtf(red[0] / 512.f) + EPSF; // std + eps
    __syncthreads();

    float sg = sigf_((score - mu) / sA);
    for (int k = 0; k < HID; ++k) xscaled[tid * HID + k] = h[tid * HID + k] * sg;
    sv[tid] = sg; si[tid] = tid;
    __syncthreads();

    for (int k = 2; k <= 512; k <<= 1)
        for (int j = k >> 1; j > 0; j >>= 1) {
            int ixj = tid ^ j;
            if (ixj > tid) {
                bool desc = ((tid & k) == 0);
                float a = sv[tid], b = sv[ixj];
                bool sw = desc ? (a < b) : (a > b);
                if (sw) { sv[tid] = b; sv[ixj] = a; int t = si[tid]; si[tid] = si[ixj]; si[ixj] = t; }
            }
            __syncthreads();
        }

    float ss = sv[tid];
    red[tid] = (tid < KSEL) ? logf(ss + EPSF) : logf(1.f - ss + EPSF);
    __syncthreads();
    for (int s = 256; s > 0; s >>= 1) { if (tid < s) red[tid] += red[tid + s]; __syncthreads(); }
    if (tid == 0) scalars[0] = -red[0] / 512.f;

    if (tid < KSEL) {
        int srcr = si[tid];
        for (int k = 0; k < HID; ++k) newx[tid * HID + k] = xscaled[srcr * HID + k];
    }
}

// high = flatten(newx) @ fc_W + fc_b   (4864 x 64)
__global__ __launch_bounds__(256)
void k_fc(const float* __restrict__ newx, const float* __restrict__ W,
          const float* __restrict__ b, float* __restrict__ high)
{
    __shared__ float red[256];
    int tid = threadIdx.x, j = tid & 63, p = tid >> 6;
    float acc = 0.f;
    for (int i = p; i < KSEL * HID; i += 4) acc += newx[i] * W[(size_t)i * HID + j];
    red[tid] = acc; __syncthreads();
    if (p == 0) high[j] = red[j] + red[64 + j] + red[128 + j] + red[192 + j] + b[j];
}

// 200-step scalar LSTM, one persistent block, one gate row per thread
__global__ __launch_bounds__(256)
void k_lstm(const float* __restrict__ ts, const float* __restrict__ Wih,
            const float* __restrict__ Whh, const float* __restrict__ bih,
            const float* __restrict__ bhh, float* __restrict__ low)
{
    __shared__ float xsh[512], hsh[64], csh[64], gsh[256];
    int tid = threadIdx.x;
    if (tid < 64) { hsh[tid] = 0.f; csh[tid] = 0.f; }
    for (int t = 0; t < TTS; ++t) {
        __syncthreads();
        xsh[tid] = ts[t * 512 + tid];
        xsh[256 + tid] = ts[t * 512 + 256 + tid];
        __syncthreads();
        float acc = bih[tid] + bhh[tid];
        const float* wr = Wih + (size_t)tid * 512;
        for (int i = 0; i < 512; ++i) acc += wr[i] * xsh[i];
        const float* wh = Whh + (size_t)tid * 64;
        for (int i = 0; i < 64; ++i) acc += wh[i] * hsh[i];
        gsh[tid] = acc;
        __syncthreads();
        if (tid < 64) {
            float ii = gsh[tid], ff = gsh[64 + tid], gg = gsh[128 + tid], oo = gsh[192 + tid];
            float c = sigf_(ff) * csh[tid] + sigf_(ii) * tanhf(gg);
            hsh[tid] = sigf_(oo) * tanhf(c); csh[tid] = c;
        }
    }
    __syncthreads();
    if (tid < 64) low[tid] = hsh[tid];
}

// fusion -> layernorm -> MLP -> pred; emit [pred, pool_loss]
__global__ __launch_bounds__(128)
void k_final(const float* __restrict__ high, const float* __restrict__ low,
             const float* __restrict__ lng, const float* __restrict__ lnb,
             const float* __restrict__ W1, const float* __restrict__ b1,
             const float* __restrict__ W2, const float* __restrict__ b2,
             const float* __restrict__ scalars, float* __restrict__ out)
{
    __shared__ float fus[128], red[128], hdn[64];
    int tid = threadIdx.x;
    float x = (tid < 64) ? high[tid] : low[tid - 64];
    red[tid] = x; __syncthreads();
    for (int s = 64; s > 0; s >>= 1) { if (tid < s) red[tid] += red[tid + s]; __syncthreads(); }
    float mu = red[0] / 128.f; __syncthreads();
    float d = x - mu;
    red[tid] = d * d; __syncthreads();
    for (int s = 64; s > 0; s >>= 1) { if (tid < s) red[tid] += red[tid + s]; __syncthreads(); }
    float var = red[0] / 128.f;
    fus[tid] = lng[tid] * d * rsqrtf(var + 1e-5f) + lnb[tid];
    __syncthreads();
    if (tid < 64) {
        float acc = b1[tid];
        for (int i = 0; i < 128; ++i) acc += fus[i] * W1[i * HID + tid];
        hdn[tid] = fmaxf(acc, 0.f);
    }
    __syncthreads();
    red[tid] = (tid < 64) ? hdn[tid] * W2[tid] : 0.f;
    __syncthreads();
    for (int s = 64; s > 0; s >>= 1) { if (tid < s) red[tid] += red[tid + s]; __syncthreads(); }
    if (tid == 0) { out[0] = red[0] + b2[0]; out[1] = scalars[0]; }
}

// ---------------------------------------------------------------------------
extern "C" void kernel_launch(void* const* d_in, const int* in_sizes, int n_in,
                              void* d_out, int out_size, void* d_ws, size_t ws_size,
                              hipStream_t stream)
{
    const float* lw   = (const float*)d_in[0];
    const int*   ei   = (const int*)d_in[1];
    const float* h0   = (const float*)d_in[2];
    const float* c0   = (const float*)d_in[3];
    const float* ts   = (const float*)d_in[4];
    const float *Wix = (const float*)d_in[5],  *bxi = (const float*)d_in[6];
    const float *Wfx = (const float*)d_in[7],  *bxf = (const float*)d_in[8];
    const float *Wox = (const float*)d_in[9],  *bxo = (const float*)d_in[10];
    const float *Wmx = (const float*)d_in[11], *bxm = (const float*)d_in[12];
    const float *Wih_ = (const float*)d_in[13], *bhi = (const float*)d_in[14];
    const float *Wfh = (const float*)d_in[15], *bhf = (const float*)d_in[16];
    const float *Woh = (const float*)d_in[17], *bho = (const float*)d_in[18];
    const float *Wmh = (const float*)d_in[19], *bhm = (const float*)d_in[20];
    const float *pv  = (const float*)d_in[21];
    const float *fcW = (const float*)d_in[22], *fcb = (const float*)d_in[23];
    const float *lWih = (const float*)d_in[24], *lWhh = (const float*)d_in[25];
    const float *lbih = (const float*)d_in[26], *lbhh = (const float*)d_in[27];
    const float *lng = (const float*)d_in[28], *lnb = (const float*)d_in[29];
    const float *m1W = (const float*)d_in[30], *m1b = (const float*)d_in[31];
    const float *m2W = (const float*)d_in[32], *m2b = (const float*)d_in[33];

    const int E = in_sizes[1] / 2;

    char* p = (char*)d_ws;
    auto alloc = [&](size_t bytes) -> char* {
        char* r = p; p += (bytes + 255) & ~(size_t)255; return r;
    };
    float*    deg  = (float*)   alloc(N_NODES * 4);
    float*    dinv = (float*)   alloc(N_NODES * 4);
    float*    Af   = (float*)   alloc((size_t)N_NODES * N_NODES * 4);
    _Float16* Ahpk = (_Float16*)alloc((size_t)N_NODES * N_NODES * 2);
    _Float16* Xpk  = (_Float16*)alloc((size_t)TSEQ * N_NODES * F_IN_ * 2);
    _Float16* Wxpk = (_Float16*)alloc((size_t)F_IN_ * GDIM * 2);
    _Float16* Whpk = (_Float16*)alloc((size_t)HID * GDIM * 2);
    _Float16* XWpk = (_Float16*)alloc((size_t)TSEQ * N_NODES * GDIM * 2);
    float*    AXW  = (float*)   alloc((size_t)TSEQ * N_NODES * GDIM * 4);
    _Float16* HWpk = (_Float16*)alloc((size_t)N_NODES * GDIM * 2);
    float*    AHW  = (float*)   alloc((size_t)N_NODES * GDIM * 4);
    float*    hf   = (float*)   alloc((size_t)N_NODES * HID * 4);
    _Float16* hhpk = (_Float16*)alloc((size_t)N_NODES * HID * 2);
    float*    cf   = (float*)   alloc((size_t)N_NODES * HID * 4);
    float*    xsc  = (float*)   alloc((size_t)N_NODES * HID * 4);
    float*    newx = (float*)   alloc((size_t)KSEL * HID * 4);
    float*    high = (float*)   alloc(HID * 4);
    float*    low  = (float*)   alloc(HID * 4);
    float*    scal = (float*)   alloc(16 * 4);

    // --- dense normalized adjacency -------------------------------------
    hipMemsetAsync(deg, 0, N_NODES * 4, stream);
    hipMemsetAsync(Af, 0, (size_t)N_NODES * N_NODES * 4, stream);
    k_deg<<<(E + 255) / 256, 256, 0, stream>>>(ei, E, deg);
    k_dinv<<<2, 256, 0, stream>>>(deg, dinv);
    k_adj<<<(E + 255) / 256, 256, 0, stream>>>(ei, E, dinv, Af);
    k_selfloop<<<2, 256, 0, stream>>>(dinv, Af);

    // --- fragment packing -----------------------------------------------
    k_packA<<<(N_NODES * F_IN_ + 255) / 256, 256, 0, stream>>>(Af, Ahpk, 1);
    k_packA<<<(TSEQ * N_NODES * F_IN_ + 255) / 256, 256, 0, stream>>>(lw, Xpk, TSEQ);
    k_pack4B<<<(F_IN_ * GDIM + 255) / 256, 256, 0, stream>>>(Wix, Wfx, Wox, Wmx, Wxpk, F_IN_);
    k_pack4B<<<(HID * GDIM + 255) / 256, 256, 0, stream>>>(Wih_, Wfh, Woh, Wmh, Whpk, HID);
    k_inith<<<(N_NODES * HID + 255) / 256, 256, 0, stream>>>(h0, c0, hf, hhpk, cf);

    const long long sNG = (long long)N_NODES * GDIM;   // 512*256
    const long long sNF = (long long)N_NODES * F_IN_;  // 512*512

    // XW[t] = X[t] @ Wx_all  (batched over 16 t) -> B-pack f16
    {
        dim3 g(GDIM / 32, N_NODES / 32, TSEQ);
        wmma_gemm_pk<<<g, 32, 0, stream>>>(Xpk, Wxpk, nullptr, XWpk,
                                           F_IN_, GDIM, N_NODES / 32, sNF, 0, sNG);
    }
    // AXW[t] = A @ XW[t]  (batched) -> f32 row-major
    {
        dim3 g(GDIM / 32, N_NODES / 32, TSEQ);
        wmma_gemm_pk<<<g, 32, 0, stream>>>(Ahpk, XWpk, AXW, nullptr,
                                           N_NODES, GDIM, 0, 0, sNG, sNG);
    }

    // --- sequential graph-LSTM recurrence -------------------------------
    for (int t = 0; t < TSEQ; ++t) {
        dim3 g(GDIM / 32, N_NODES / 32, 1);
        // HW = h (512x64) @ Wh_all (64x256) -> B-pack f16
        wmma_gemm_pk<<<g, 32, 0, stream>>>(hhpk, Whpk, nullptr, HWpk,
                                           HID, GDIM, N_NODES / 32, 0, 0, 0);
        // AHW = A @ HW -> f32 row-major
        wmma_gemm_pk<<<g, 32, 0, stream>>>(Ahpk, HWpk, AHW, nullptr,
                                           N_NODES, GDIM, 0, 0, 0, 0);
        k_gate<<<(N_NODES * HID) / 256, 256, 0, stream>>>(
            AXW + (size_t)t * sNG, AHW,
            bxi, bxf, bxo, bxm, bhi, bhf, bho, bhm, cf, hf, hhpk);
    }

    // --- pooling, heads --------------------------------------------------
    k_pool<<<1, 512, 0, stream>>>(hf, pv, xsc, newx, scal);
    k_fc<<<1, 256, 0, stream>>>(newx, fcW, fcb, high);
    k_lstm<<<1, 256, 0, stream>>>(ts, lWih, lWhh, lbih, lbhh, low);
    k_final<<<1, 128, 0, stream>>>(high, low, lng, lnb, m1W, m1b, m2W, m2b,
                                   scal, (float*)d_out);
}
